// Predictor_seq_14757507629580
// MI455X (gfx1250) — matrix operand order repaired and checked by reading, not compiled, not run on hardware
//
#include <hip/hip_runtime.h>
#include <hip/hip_bf16.h>
#include <stdint.h>

#define B_SZ   32768
#define NPRED  25
#define BN_EPS 1e-5f
#define MT     2                 // M-tiles per wave (32 batch rows)
#define RPW    (MT * 16)         // rows per wave
#define RPB    (8 * RPW)         // rows per block (8 waves)

typedef __attribute__((ext_vector_type(16))) __bf16 v16bf;
typedef __attribute__((ext_vector_type(8)))  float  v8f;

union BFV {                      // 16 bf16 = 32 bytes = two 16B chunks
    v16bf v;
    uint4 q[2];
};

// CDNA5 hardware TANH transcendental (ISA §8.4). Gates fold their bias into
// the nonlinearity: sigmoid(x+b) = 0.5 + 0.5*tanh(0.5x + 0.5b)  (hb = 0.5b,
// the 0.5x mul becomes an fma -> bias is free);  tanh(x+b) costs one add.
#if __has_builtin(__builtin_amdgcn_tanhf)
__device__ __forceinline__ float gate_sig(float x, float hb) {
    return 0.5f + 0.5f * __builtin_amdgcn_tanhf(fmaf(0.5f, x, hb));
}
__device__ __forceinline__ float gate_tanh(float x, float b) {
    return __builtin_amdgcn_tanhf(x + b);
}
__device__ __forceinline__ float tanh_f(float x) {
    return __builtin_amdgcn_tanhf(x);
}
#else
__device__ __forceinline__ float gate_sig(float x, float hb) {
    return 1.0f / (1.0f + __expf(-x - 2.0f * hb));
}
__device__ __forceinline__ float gate_tanh(float x, float b) {
    return 1.0f - 2.0f / (__expf(2.0f * (x + b)) + 1.0f);
}
__device__ __forceinline__ float tanh_f(float x) {
    return 1.0f - 2.0f / (__expf(2.0f * x) + 1.0f);
}
#endif

// A-fragment (16x32 bf16, contiguous k = 16*half + s): lane row = lrow, K split by half
__device__ __forceinline__ v16bf load_afrag(const __bf16* rowp, int kb, int half) {
    BFV t;
    int o = kb * 32 + half * 16;
    t.q[0] = *(const uint4*)(rowp + o);
    t.q[1] = *(const uint4*)(rowp + o + 8);
    return t.v;
}
// B-fragment (32x16 bf16): lane col = lrow, contiguous K run per lane
__device__ __forceinline__ v16bf load_bfrag(const __bf16* W, int nrow, int kb, int half) {
    BFV t;
    const __bf16* wp = W + nrow * 64 + kb * 32 + half * 16;
    t.q[0] = *(const uint4*)(wp);
    t.q[1] = *(const uint4*)(wp + 8);
    return t.v;
}

// ---------------------------------------------------------------------------
// stats zeroing (BN stats accumulated with atomics each call)
// ---------------------------------------------------------------------------
__global__ void zero_stats_kernel(float* stats) {
    int i = threadIdx.x;
    if (i < 384) stats[i] = 0.0f;
}

// ---------------------------------------------------------------------------
// MLP layer 1: h1 = x @ W1^T + b1  (64 -> 32), accumulate per-frame stats
// ---------------------------------------------------------------------------
__global__ __launch_bounds__(256) void mlp1_kernel(
    const float* __restrict__ x,    // [3][B][64]
    const float* __restrict__ W1,   // [32][64]
    const float* __restrict__ b1,   // [32]
    float* __restrict__ h1,         // [3][B][32]
    float* __restrict__ stats1)     // [3][64]  (sum[32], sumsq[32])
{
    __shared__ float sW[32 * 64];
    __shared__ float sb[32], ssum[32], ssq[32];
    int tid = threadIdx.x;
    for (int i = tid; i < 32 * 64; i += 256) sW[i] = W1[i];
    if (tid < 32) { sb[tid] = b1[tid]; ssum[tid] = 0.0f; ssq[tid] = 0.0f; }
    __syncthreads();

    int f = blockIdx.y;
    int r = blockIdx.x * 256 + tid;
    int lane = tid & 31;
    const float* xr = x + ((size_t)f * B_SZ + r) * 64;
    float xv[64];
#pragma unroll
    for (int k = 0; k < 64; ++k) xv[k] = xr[k];

    float* ho = h1 + ((size_t)f * B_SZ + r) * 32;
    for (int j = 0; j < 32; ++j) {
        float a = sb[j];
#pragma unroll
        for (int k = 0; k < 64; ++k) a += xv[k] * sW[j * 64 + k];
        ho[j] = a;
        float s1 = a, s2 = a * a;
#pragma unroll
        for (int off = 16; off; off >>= 1) {
            s1 += __shfl_xor(s1, off, 32);
            s2 += __shfl_xor(s2, off, 32);
        }
        if (lane == 0) { atomicAdd(&ssum[j], s1); atomicAdd(&ssq[j], s2); }
    }
    __syncthreads();
    if (tid < 32) {
        atomicAdd(&stats1[f * 64 + tid],      ssum[tid]);
        atomicAdd(&stats1[f * 64 + 32 + tid], ssq[tid]);
    }
}

// ---------------------------------------------------------------------------
// MLP layer 2: a = relu(BN1(h1)); h2 = a @ W2^T + b2  (32 -> 32), + stats
// ---------------------------------------------------------------------------
__global__ __launch_bounds__(256) void mlp2_kernel(
    const float* __restrict__ h1, const float* __restrict__ stats1,
    const float* __restrict__ gam, const float* __restrict__ bet,
    const float* __restrict__ W2, const float* __restrict__ b2,
    float* __restrict__ h2, float* __restrict__ stats2)
{
    __shared__ float sW[32 * 32];
    __shared__ float sb[32], sg[32], sbe[32], smean[32], srstd[32];
    __shared__ float ssum[32], ssq[32];
    int tid = threadIdx.x, f = blockIdx.y;
    for (int i = tid; i < 32 * 32; i += 256) sW[i] = W2[i];
    if (tid < 32) {
        sb[tid] = b2[tid]; sg[tid] = gam[tid]; sbe[tid] = bet[tid];
        float m   = stats1[f * 64 + tid] * (1.0f / (float)B_SZ);
        float var = stats1[f * 64 + 32 + tid] * (1.0f / (float)B_SZ) - m * m;
        smean[tid] = m; srstd[tid] = rsqrtf(var + BN_EPS);
        ssum[tid] = 0.0f; ssq[tid] = 0.0f;
    }
    __syncthreads();

    int r = blockIdx.x * 256 + tid;
    int lane = tid & 31;
    const float* hr = h1 + ((size_t)f * B_SZ + r) * 32;
    float a[32];
#pragma unroll
    for (int k = 0; k < 32; ++k) {
        float t = sg[k] * (hr[k] - smean[k]) * srstd[k] + sbe[k];
        a[k] = t > 0.0f ? t : 0.0f;
    }
    float* ho = h2 + ((size_t)f * B_SZ + r) * 32;
    for (int j = 0; j < 32; ++j) {
        float acc = sb[j];
#pragma unroll
        for (int k = 0; k < 32; ++k) acc += a[k] * sW[j * 32 + k];
        ho[j] = acc;
        float s1 = acc, s2 = acc * acc;
#pragma unroll
        for (int off = 16; off; off >>= 1) {
            s1 += __shfl_xor(s1, off, 32);
            s2 += __shfl_xor(s2, off, 32);
        }
        if (lane == 0) { atomicAdd(&ssum[j], s1); atomicAdd(&ssq[j], s2); }
    }
    __syncthreads();
    if (tid < 32) {
        atomicAdd(&stats2[f * 64 + tid],      ssum[tid]);
        atomicAdd(&stats2[f * 64 + 32 + tid], ssq[tid]);
    }
}

// ---------------------------------------------------------------------------
// MLP layer 3: a = relu(BN2(h2)); enc = a @ W3^T + b3  (32 -> 64), store bf16
// ---------------------------------------------------------------------------
__global__ __launch_bounds__(256) void mlp3_kernel(
    const float* __restrict__ h2, const float* __restrict__ stats2,
    const float* __restrict__ gam, const float* __restrict__ bet,
    const float* __restrict__ W3, const float* __restrict__ b3,
    __bf16* __restrict__ enc)       // [3][B][64] bf16
{
    __shared__ float sW[64 * 32];
    __shared__ float sb[64], sg[32], sbe[32], smean[32], srstd[32];
    int tid = threadIdx.x, f = blockIdx.y;
    for (int i = tid; i < 64 * 32; i += 256) sW[i] = W3[i];
    if (tid < 64) sb[tid] = b3[tid];
    if (tid < 32) {
        sg[tid] = gam[tid]; sbe[tid] = bet[tid];
        float m   = stats2[f * 64 + tid] * (1.0f / (float)B_SZ);
        float var = stats2[f * 64 + 32 + tid] * (1.0f / (float)B_SZ) - m * m;
        smean[tid] = m; srstd[tid] = rsqrtf(var + BN_EPS);
    }
    __syncthreads();

    int r = blockIdx.x * 256 + tid;
    const float* hr = h2 + ((size_t)f * B_SZ + r) * 32;
    float a[32];
#pragma unroll
    for (int k = 0; k < 32; ++k) {
        float t = sg[k] * (hr[k] - smean[k]) * srstd[k] + sbe[k];
        a[k] = t > 0.0f ? t : 0.0f;
    }
    __bf16* eo = enc + ((size_t)f * B_SZ + r) * 64;
    for (int j = 0; j < 64; ++j) {
        float acc = sb[j];
#pragma unroll
        for (int k = 0; k < 32; ++k) acc += a[k] * sW[j * 32 + k];
        eo[j] = (__bf16)acc;
    }
}

// ---------------------------------------------------------------------------
// Lane-local LSTM cell update for one (m, jt) tile. Gate accumulators are
// bias-free (WMMA chains start from C=0); bias is folded in here.
// bI/bF/bO are 0.5*bias (sigmoid gates), bG is the full bias (tanh gate).
// ---------------------------------------------------------------------------
template<bool WRITE>
__device__ __forceinline__ void cell_update(
    const v8f (&acc)[4], float (&cvec)[8],
    float bI, float bF, float bG, float bO,
    __bf16* hbuf, float* __restrict__ out, int outStep,
    int rowoff /* m*16 + half*8 */, int col, int rowbase)
{
#pragma unroll
    for (int v = 0; v < 8; ++v) {
        float iv = gate_sig(acc[0][v], bI);
        float fv = gate_sig(acc[1][v], bF);
        float gv = gate_tanh(acc[2][v], bG);
        float ov = gate_sig(acc[3][v], bO);
        float c  = fv * cvec[v] + iv * gv;
        cvec[v]  = c;
        float h  = ov * tanh_f(c);
        int row  = rowoff + v;
        hbuf[row * 64 + col] = (__bf16)h;
        if (WRITE)
            out[((size_t)(rowbase + row)) * (NPRED * 64)
                + (size_t)outStep * 64 + col] = h;
    }
}

// ---------------------------------------------------------------------------
// Conditioning-phase step (steps 0..3): gates = enc@Wih^T (+ h@Whh^T) + bias.
// Processes one M-tile at a time to keep VGPR pressure low.
// ---------------------------------------------------------------------------
template<bool HASH, bool WRITE>
__device__ __forceinline__ void lstm_step_cond(
    const __bf16* __restrict__ enc, int encIdx,
    const __bf16* __restrict__ Wih, const __bf16* __restrict__ Whh,
    const float (&biasreg)[16], float (&cst)[MT][4][8],
    __bf16* hbuf, float* __restrict__ out,
    int rowbase, int lrow, int half)
{
#pragma unroll
    for (int m = 0; m < MT; ++m) {
        v16bf xA[2], hA[2];
        const __bf16* ep =
            enc + ((size_t)encIdx * B_SZ + rowbase + m * 16 + lrow) * 64;
#pragma unroll
        for (int kb = 0; kb < 2; ++kb) xA[kb] = load_afrag(ep, kb, half);
        if (HASH) {
            const __bf16* hp = hbuf + (m * 16 + lrow) * 64;
#pragma unroll
            for (int kb = 0; kb < 2; ++kb) hA[kb] = load_afrag(hp, kb, half);
        }
#pragma unroll
        for (int jt = 0; jt < 4; ++jt) {
            v8f acc[4];
#pragma unroll
            for (int g4 = 0; g4 < 4; ++g4) {
                int nrow = (g4 * 4 + jt) * 16 + lrow;
                v8f a = (v8f){0.f, 0.f, 0.f, 0.f, 0.f, 0.f, 0.f, 0.f};
#pragma unroll
                for (int kb = 0; kb < 2; ++kb) {
                    v16bf bx = load_bfrag(Wih, nrow, kb, half);
                    a = __builtin_amdgcn_wmma_f32_16x16x32_bf16(
                            false, xA[kb], false, bx, (short)0, a, false, false);
                }
                if (HASH) {
#pragma unroll
                    for (int kb = 0; kb < 2; ++kb) {
                        v16bf bh = load_bfrag(Whh, nrow, kb, half);
                        a = __builtin_amdgcn_wmma_f32_16x16x32_bf16(
                                false, hA[kb], false, bh, (short)0, a, false, false);
                    }
                }
                acc[g4] = a;
            }
            cell_update<WRITE>(acc, cst[m][jt],
                               biasreg[jt], biasreg[4 + jt],
                               biasreg[8 + jt], biasreg[12 + jt],
                               hbuf, out, 0,
                               m * 16 + half * 8, jt * 16 + lrow, rowbase);
        }
    }
}

// ---------------------------------------------------------------------------
// Prediction-phase step (steps 4..27): inp == h, gates = h @ Wsum^T + bias.
// MT=2 row-tiles share each B fragment -> 2 independent WMMA chains per load.
// ---------------------------------------------------------------------------
__device__ __forceinline__ void lstm_step_pred(
    const __bf16* __restrict__ Wsum,
    const float (&biasreg)[16], float (&cst)[MT][4][8],
    __bf16* hbuf, float* __restrict__ out, int outStep,
    int rowbase, int lrow, int half)
{
    v16bf hA[MT][2];
#pragma unroll
    for (int m = 0; m < MT; ++m) {
        const __bf16* hp = hbuf + (m * 16 + lrow) * 64;
#pragma unroll
        for (int kb = 0; kb < 2; ++kb) hA[m][kb] = load_afrag(hp, kb, half);
    }
#pragma unroll
    for (int jt = 0; jt < 4; ++jt) {
        v8f acc[MT][4];
#pragma unroll
        for (int g4 = 0; g4 < 4; ++g4) {
            int nrow = (g4 * 4 + jt) * 16 + lrow;
            v16bf b0 = load_bfrag(Wsum, nrow, 0, half);
            v16bf b1 = load_bfrag(Wsum, nrow, 1, half);
#pragma unroll
            for (int m = 0; m < MT; ++m) {
                v8f a = (v8f){0.f, 0.f, 0.f, 0.f, 0.f, 0.f, 0.f, 0.f};
                a = __builtin_amdgcn_wmma_f32_16x16x32_bf16(
                        false, hA[m][0], false, b0, (short)0, a, false, false);
                a = __builtin_amdgcn_wmma_f32_16x16x32_bf16(
                        false, hA[m][1], false, b1, (short)0, a, false, false);
                acc[m][g4] = a;
            }
        }
#pragma unroll
        for (int m = 0; m < MT; ++m)
            cell_update<true>(acc[m], cst[m][jt],
                              biasreg[jt], biasreg[4 + jt],
                              biasreg[8 + jt], biasreg[12 + jt],
                              hbuf, out, outStep,
                              m * 16 + half * 8, jt * 16 + lrow, rowbase);
    }
}

// ---------------------------------------------------------------------------
// LSTM: 28 sequential steps, batch-parallel. One wave = 32 batch rows.
// Weights (bf16) + fused bias in LDS; h tiles in per-wave LDS; c in VGPRs.
// ---------------------------------------------------------------------------
__global__ __launch_bounds__(256) void lstm_wmma_kernel(
    const __bf16* __restrict__ enc,  // [3][B][64] bf16
    const float* __restrict__ Wih,   // [256][64]
    const float* __restrict__ Whh,   // [256][64]
    const float* __restrict__ bih, const float* __restrict__ bhh,
    float* __restrict__ out)         // [B][25][64]
{
    extern __shared__ char smem[];
    __bf16* sWih  = (__bf16*)smem;              // 32KB
    __bf16* sWhh  = sWih  + 256 * 64;           // 32KB
    __bf16* sWsum = sWhh  + 256 * 64;           // 32KB
    float*  sBias = (float*)(sWsum + 256 * 64); // 1KB
    __bf16* sH    = (__bf16*)(sBias + 256);     // 8 * 32*64 bf16 = 32KB

    int tid = threadIdx.x;
    for (int i = tid; i < 256 * 64; i += 256) {
        float a = Wih[i], b = Whh[i];
        sWih[i]  = (__bf16)a;
        sWhh[i]  = (__bf16)b;
        sWsum[i] = (__bf16)(a + b);
    }
    if (tid < 256) sBias[tid] = bih[tid] + bhh[tid];
    __syncthreads();

    int wave = tid >> 5;
    int lane = tid & 31;
    int lrow = lane & 15;
    int half = lane >> 4;
    int rowbase = blockIdx.x * RPB + wave * RPW;
    __bf16* hbuf = sH + wave * (RPW * 64);

    // Pre-scaled per-tile bias: 0.5*b for sigmoid gates (i,f,o), full b for g.
    float biasreg[16];
#pragma unroll
    for (int nt = 0; nt < 16; ++nt) {
        float raw = sBias[nt * 16 + lrow];
        biasreg[nt] = ((nt >> 2) == 2) ? raw : 0.5f * raw;
    }

    float cst[MT][4][8];
#pragma unroll
    for (int m = 0; m < MT; ++m)
#pragma unroll
        for (int jt = 0; jt < 4; ++jt)
#pragma unroll
            for (int v = 0; v < 8; ++v) cst[m][jt][v] = 0.0f;

    // ---- step 0: gates = enc0 @ Wih^T + bias (h = c = 0) ----
    lstm_step_cond<false, false>(enc, 0, sWih, sWhh, biasreg, cst,
                                 hbuf, out, rowbase, lrow, half);
    // ---- steps 1,2: gates = enc[s] @ Wih^T + h @ Whh^T + bias ----
    for (int s = 1; s <= 2; ++s)
        lstm_step_cond<true, false>(enc, s, sWih, sWhh, biasreg, cst,
                                    hbuf, out, rowbase, lrow, half);
    // ---- step 3 (pred 0): inp = enc2, h = h3; writes output index 0 ----
    lstm_step_cond<true, true>(enc, 2, sWih, sWhh, biasreg, cst,
                               hbuf, out, rowbase, lrow, half);
    // ---- steps 4..27 (pred 1..24): inp == h -> single GEMM with Wsum ----
    for (int s = 4; s < 28; ++s)
        lstm_step_pred(sWsum, biasreg, cst, hbuf, out, s - 3,
                       rowbase, lrow, half);
}

// ---------------------------------------------------------------------------
extern "C" void kernel_launch(void* const* d_in, const int* in_sizes, int n_in,
                              void* d_out, int out_size, void* d_ws, size_t ws_size,
                              hipStream_t stream) {
    const float* x    = (const float*)d_in[0];
    const float* W1   = (const float*)d_in[1];
    const float* b1   = (const float*)d_in[2];
    const float* g1v  = (const float*)d_in[3];
    const float* be1  = (const float*)d_in[4];
    const float* W2   = (const float*)d_in[5];
    const float* b2   = (const float*)d_in[6];
    const float* g2v  = (const float*)d_in[7];
    const float* be2  = (const float*)d_in[8];
    const float* W3   = (const float*)d_in[9];
    const float* b3   = (const float*)d_in[10];
    const float* Wih  = (const float*)d_in[11];
    const float* Whh  = (const float*)d_in[12];
    const float* bih  = (const float*)d_in[13];
    const float* bhh  = (const float*)d_in[14];
    float* out = (float*)d_out;

    char* ws = (char*)d_ws;
    float*  stats1 = (float*)ws;                             // [3][64]
    float*  stats2 = stats1 + 192;                           // [3][64]
    float*  h1     = (float*)(ws + 4096);                    // [3][B][32]
    float*  h2     = h1 + (size_t)3 * B_SZ * 32;             // [3][B][32]
    __bf16* enc    = (__bf16*)(h2 + (size_t)3 * B_SZ * 32);  // [3][B][64] bf16

    zero_stats_kernel<<<1, 384, 0, stream>>>(stats1);

    dim3 mg(B_SZ / 256, 3);
    mlp1_kernel<<<mg, 256, 0, stream>>>(x, W1, b1, h1, stats1);
    mlp2_kernel<<<mg, 256, 0, stream>>>(h1, stats1, g1v, be1, W2, b2, h2, stats2);
    mlp3_kernel<<<mg, 256, 0, stream>>>(h2, stats2, g2v, be2, W3, b3, enc);

    size_t lds_bytes = (size_t)3 * 256 * 64 * 2   // Wih, Whh, Wsum (bf16)
                     + 256 * 4                    // fused bias
                     + (size_t)8 * RPW * 64 * 2;  // per-wave h tiles
    lstm_wmma_kernel<<<B_SZ / RPB, 256, lds_bytes, stream>>>(
        enc, Wih, Whh, bih, bhh, out);
}
